// Accuracy_77257871721096
// MI455X (gfx1250) — compile-verified, matrix operand order checked by reading, not run on hardware
//
#include <hip/hip_runtime.h>

typedef __attribute__((ext_vector_type(8))) int v8i;

#define N_SEQ 128
#define T_LEN 2048
#define D_FEAT 88

// Pack 4 (output,target) float pairs into 4 iu8 bytes of AND and XOR words.
// pred = output > 0 (== sigmoid(output) > 0.5); tgt = target > 0.5 (binary 0/1 floats)
__device__ __forceinline__ void pack4(float o0, float o1, float o2, float o3,
                                      float t0, float t1, float t2, float t3,
                                      unsigned& aw, unsigned& xw) {
    unsigned p0 = o0 > 0.0f, p1 = o1 > 0.0f, p2 = o2 > 0.0f, p3 = o3 > 0.0f;
    unsigned g0 = t0 > 0.5f, g1 = t1 > 0.5f, g2 = t2 > 0.5f, g3 = t3 > 0.5f;
    aw = (p0 & g0) | ((p1 & g1) << 8) | ((p2 & g2) << 16) | ((p3 & g3) << 24);
    xw = (p0 ^ g0) | ((p1 ^ g1) << 8) | ((p2 ^ g2) << 16) | ((p3 ^ g3) << 24);
}

// Pass 1: one wave handles 16 consecutive timesteps (rows) of one sequence n.
// For each row, D-axis (88-feature) reductions are done with two accumulating
// V_WMMA_I32_16X16X64_IU8 ops (K padded 88 -> 128, pad bytes are zero):
//   A = (pred & target) bytes, B = ones  ->  D[i][*] = tru_pos count of row i
//   A = (pred ^ target) bytes, B = ones  ->  D[i][*] = (fp+fn) count of row i
// B = all-ones makes the result independent of the B-matrix VGPR layout.
__global__ void __launch_bounds__(256)
accuracy_pass1(const float* __restrict__ out_t, const float* __restrict__ tgt_t,
               int* __restrict__ tp, int* __restrict__ fpfn) {
    const int lane = threadIdx.x & 31;
    const int wave = blockIdx.x * (blockDim.x >> 5) + (threadIdx.x >> 5);
    const int row_base = wave * 16;              // 16 rows (n,t pairs) per wave
    const int M = lane & 15;                     // matrix row handled by this lane
    const int h = lane >> 4;                     // half-wave (K-group select)
    const long row = (long)row_base + M;

    const float* __restrict__ orow = out_t + row * D_FEAT;
    const float* __restrict__ trow = tgt_t + row * D_FEAT;

    // Build A matrices for two K-chunks (K 0..63 and 64..127).
    // ISA 8-bit A 16x64 layout: lane l holds row l%16; VGPR 2b   = K 16b+8h+0..3,
    //                                                  VGPR 2b+1 = K 16b+8h+4..7 (LSB = lowest K).
    v8i a_and[2], a_xor[2];
#pragma unroll
    for (int c = 0; c < 2; ++c) {
#pragma unroll
        for (int b = 0; b < 4; ++b) {
            unsigned aw0 = 0, xw0 = 0, aw1 = 0, xw1 = 0;
            const int fbase = c * 64 + 16 * b + 8 * h;
            if (fbase + 8 <= D_FEAT) {           // zero-pad features 88..127
                float4 o0 = *(const float4*)(orow + fbase);
                float4 o1 = *(const float4*)(orow + fbase + 4);
                float4 t0 = *(const float4*)(trow + fbase);
                float4 t1 = *(const float4*)(trow + fbase + 4);
                pack4(o0.x, o0.y, o0.z, o0.w, t0.x, t0.y, t0.z, t0.w, aw0, xw0);
                pack4(o1.x, o1.y, o1.z, o1.w, t1.x, t1.y, t1.z, t1.w, aw1, xw1);
            }
            a_and[c][2 * b]     = (int)aw0;
            a_and[c][2 * b + 1] = (int)aw1;
            a_xor[c][2 * b]     = (int)xw0;
            a_xor[c][2 * b + 1] = (int)xw1;
        }
    }

    const v8i bones = {0x01010101, 0x01010101, 0x01010101, 0x01010101,
                       0x01010101, 0x01010101, 0x01010101, 0x01010101};
    v8i acc_tp = {0, 0, 0, 0, 0, 0, 0, 0};
    v8i acc_fx = {0, 0, 0, 0, 0, 0, 0, 0};

    // (sgn_a, A, sgn_b, B, C, reuse_a, reuse_b) -> v_wmma_i32_16x16x64_iu8
    acc_tp = __builtin_amdgcn_wmma_i32_16x16x64_iu8(false, a_and[0], false, bones, acc_tp, false, false);
    acc_tp = __builtin_amdgcn_wmma_i32_16x16x64_iu8(false, a_and[1], false, bones, acc_tp, false, false);
    acc_fx = __builtin_amdgcn_wmma_i32_16x16x64_iu8(false, a_xor[0], false, bones, acc_fx, false, false);
    acc_fx = __builtin_amdgcn_wmma_i32_16x16x64_iu8(false, a_xor[1], false, bones, acc_fx, false, false);

    // C/D layout: VGPR r, lanes 0-15 -> M=r (N=lane); lanes 16-31 -> M=r+8.
    // Row sums are replicated across all N, so lanes with N==0 (lanes 0 and 16)
    // hold results for rows 0..7 (lane 0) and 8..15 (lane 16) in their 8 VGPRs.
    if (M == 0) {
        const int n  = row_base / T_LEN;
        const int t0 = (row_base % T_LEN) + 8 * h;
        int tp_sum = 0;
#pragma unroll
        for (int r = 0; r < 8; ++r) {
            tp_sum += acc_tp[r];
            fpfn[n * T_LEN + t0 + r] = acc_fx[r];
        }
        atomicAdd(&tp[n], tp_sum);   // one atomic per half-wave; same n for all 8 rows
    }
}

// Pass 2: one block per sequence n. T_i = sum(mask row); acc = sum_t<Ti tp/(tp+fpfn);
// out += acc / T_i.
__global__ void __launch_bounds__(256)
accuracy_pass2(const int* __restrict__ mask, const int* __restrict__ tp,
               const int* __restrict__ fpfn, float* __restrict__ out) {
    const int n = blockIdx.x;
    const int tid = threadIdx.x;

    __shared__ int   s_i[256];
    __shared__ float s_f[256];

    int ti_local = 0;
    for (int t = tid; t < T_LEN; t += blockDim.x) ti_local += mask[n * T_LEN + t];
    s_i[tid] = ti_local;
    __syncthreads();
    for (int s = 128; s > 0; s >>= 1) {
        if (tid < s) s_i[tid] += s_i[tid + s];
        __syncthreads();
    }
    const int Ti = s_i[0];
    __syncthreads();

    const float tpf = (float)tp[n];
    float acc = 0.0f;
    for (int t = tid; t < T_LEN; t += blockDim.x) {
        if (t < Ti) {
            const float denom = tpf + (float)fpfn[n * T_LEN + t];
            acc += tpf / denom;
        }
    }
    s_f[tid] = acc;
    __syncthreads();
    for (int s = 128; s > 0; s >>= 1) {
        if (tid < s) s_f[tid] += s_f[tid + s];
        __syncthreads();
    }
    if (tid == 0) atomicAdd(out, s_f[0] / (float)Ti);
}

extern "C" void kernel_launch(void* const* d_in, const int* in_sizes, int n_in,
                              void* d_out, int out_size, void* d_ws, size_t ws_size,
                              hipStream_t stream) {
    const float* output = (const float*)d_in[0];   // [N,T,D] f32
    const float* target = (const float*)d_in[1];   // [N,T,D] f32 (0/1)
    const int*   mask   = (const int*)d_in[2];     // [N,T] i32
    float* out = (float*)d_out;                    // scalar

    int* tp   = (int*)d_ws;                        // [N]   tru_pos counts
    int* fpfn = tp + N_SEQ;                        // [N,T] fp+fn counts (~1 MB)

    hipMemsetAsync(tp, 0, N_SEQ * sizeof(int), stream);
    hipMemsetAsync(out, 0, sizeof(float), stream);

    // 16 rows/wave, 8 waves/block -> 128 rows/block; N*T = 262144 rows -> 2048 blocks
    const int blocks = (N_SEQ * T_LEN) / (16 * 8);
    accuracy_pass1<<<blocks, 256, 0, stream>>>(output, target, tp, fpfn);
    accuracy_pass2<<<N_SEQ, 256, 0, stream>>>(mask, tp, fpfn, out);
}
